// CovarianceDecoder_42932493091201
// MI455X (gfx1250) — matrix operand order repaired and checked by reading, not compiled, not run on hardware
//
#include <hip/hip_runtime.h>
#include <hip/hip_bf16.h>
#include <math.h>

#define D_MODEL    256
#define NUM_ASSETS 100
#define HIDDEN     256
#define NPAIR      4950     // 100*99/2
#define BATCH      64
#define TILES_PB   ((NPAIR + 15) / 16)   // 310
#define LDA        101                   // padded LDS leading dim
#define NSWEEP     10
#define KTILES     (D_MODEL / 32)        // 8
#define NTILES     (HIDDEN / 16)         // 16

typedef __attribute__((ext_vector_type(16))) __bf16 v16bf;
typedef __attribute__((ext_vector_type(8)))  __bf16 bf16x8;
typedef __attribute__((ext_vector_type(2)))  __bf16 bf16x2;
typedef __attribute__((ext_vector_type(8)))  float  v8f;

// ---------------------------------------------------------------- w1 -> bf16 fragments
// Emit corr_w1 in per-lane WMMA B-fragment order:
//   w1f[((nt*KTILES + kt)*32 + lane)*16 + e] = w1[(kt*32 + half*8 + (e&7) + (e>>3)*16)*HIDDEN + nt*16 + n]
// so the GEMM loads one contiguous 32B v16bf per (lane, fragment).
__global__ __launch_bounds__(256)
void cvt_w1_kernel(const float* __restrict__ w1, __bf16* __restrict__ w1f) {
  int idx = blockIdx.x * 256 + threadIdx.x;   // 0 .. 65535
  int e    = idx & 15;
  int lane = (idx >> 4) & 31;
  int kt   = (idx >> 9) & 7;
  int nt   = idx >> 12;
  int half = lane >> 4;
  int n    = lane & 15;
  w1f[idx] = (__bf16)w1[(kt * 32 + half * 8 + (e & 7) + ((e >> 3) << 4)) * HIDDEN + nt * 16 + n];
}

// ---------------------------------------------------------------- vol MLP
__global__ __launch_bounds__(256)
void vol_kernel(const float* __restrict__ x,
                const float* __restrict__ w1, const float* __restrict__ b1,
                const float* __restrict__ w2, const float* __restrict__ b2,
                float* __restrict__ vols) {
  __shared__ float row[D_MODEL];
  __shared__ float red[HIDDEN];
  int blk = blockIdx.x;
  int b = blk / NUM_ASSETS;
  int a = blk - b * NUM_ASSETS;
  int t = threadIdx.x;
  const float* xr = x + (((size_t)b * 64 + 63) * NUM_ASSETS + a) * D_MODEL;
  row[t] = xr[t];
  __syncthreads();
  float h = 0.f;
  #pragma unroll 8
  for (int d = 0; d < D_MODEL; ++d) h = fmaf(row[d], w1[d * HIDDEN + t], h);
  red[t] = fmaxf(h + b1[t], 0.f) * w2[t];
  __syncthreads();
  for (int off = 128; off > 0; off >>= 1) {
    if (t < off) red[t] += red[t + off];
    __syncthreads();
  }
  if (t == 0) {
    float raw = red[0] + b2[0];
    float sp  = (raw > 20.f) ? raw : log1pf(expf(raw));   // softplus
    vols[blk] = sp + 1e-6f;
  }
}

// ---------------------------------------------------------------- corr MLP (WMMA)
__global__ __launch_bounds__(256)
void corr_kernel(const float* __restrict__ x,
                 const __bf16* __restrict__ w1f, const float* __restrict__ b1,
                 const float* __restrict__ w2, const float* __restrict__ b2,
                 float* __restrict__ tril) {
  __shared__ __align__(16) __bf16 Abuf[16][D_MODEL];
  __shared__ float  rowsum[16];
  __shared__ int    tiL[16], tjL[16];

  int blk  = blockIdx.x;
  int b    = blk / TILES_PB;
  int tile = blk - b * TILES_PB;
  int tid  = threadIdx.x;
  int p0   = tile * 16;

  if (tid < 16) {
    rowsum[tid] = 0.f;
    int p = p0 + tid;
    if (p >= NPAIR) p = 0;
    // decode flat tril index -> (i, j), i > j (np.tril_indices order)
    int i = (int)((1.0f + sqrtf(1.0f + 8.0f * (float)p)) * 0.5f);
    while (i * (i - 1) / 2 > p) --i;
    while ((i + 1) * i / 2 <= p) ++i;
    tiL[tid] = i;
    tjL[tid] = p - i * (i - 1) / 2;
  }
  __syncthreads();

  // stage 16 pair rows (emb_i * emb_j) into LDS as bf16 (float2 loads, b32 stores)
  const float* xb = x + (((size_t)b * 64 + 63) * NUM_ASSETS) * D_MODEL;
  {
    int rbase = (tid >> 7) * 8;        // rows 0..7 or 8..15
    int cp    = (tid & 127) << 1;      // even column pair
    #pragma unroll
    for (int r8 = 0; r8 < 8; ++r8) {
      int row = rbase + r8;
      const float2 a2 = *(const float2*)&xb[tiL[row] * D_MODEL + cp];
      const float2 b2v = *(const float2*)&xb[tjL[row] * D_MODEL + cp];
      bf16x2 pk;
      pk[0] = (__bf16)(a2.x * b2v.x);
      pk[1] = (__bf16)(a2.y * b2v.y);
      *(bf16x2*)&Abuf[row][cp] = pk;
    }
  }
  __syncthreads();

  int wave = tid >> 5, lane = tid & 31;
  int m = lane & 15, half = lane >> 4;
  const v16bf* w1v = (const v16bf*)w1f;
  const bf16x8* arow = (const bf16x8*)&Abuf[m][0];

  #pragma unroll
  for (int ni = 0; ni < 2; ++ni) {
    int nt = wave + ni * 8;
    v8f acc = {};
    #pragma unroll
    for (int kt = 0; kt < KTILES; ++kt) {
      // A fragment: two contiguous 16B runs of the LDS row -> 2x ds_load_b128
      bf16x8 lo = arow[(kt << 2) + half];          // k0 + half*8 .. +7
      bf16x8 hi = arow[(kt << 2) + 2 + half];      // k0 + 16 + half*8 .. +7
      v16bf af = __builtin_shufflevector(lo, hi,
          0, 1, 2, 3, 4, 5, 6, 7, 8, 9, 10, 11, 12, 13, 14, 15);
      // B fragment: one contiguous 32B per lane -> 2x global_load_b128
      v16bf bfm = w1v[(nt * KTILES + kt) * 32 + lane];
      acc = __builtin_amdgcn_wmma_f32_16x16x32_bf16(
          false, af, false, bfm, (short)0, acc, false, false);
    }
    // layer-2 collapse: relu(h + b1) * w2, reduce over hidden columns
    int   colh = nt * 16 + m;
    float bias = b1[colh];
    float wv   = w2[colh];
    #pragma unroll
    for (int v = 0; v < 8; ++v) {
      float val = fmaxf(acc[v] + bias, 0.f) * wv;
      #pragma unroll
      for (int off = 1; off < 16; off <<= 1)
        val += __shfl_xor(val, off, 16);                   // reduce 16 cols
      if (m == 0) atomicAdd(&rowsum[half * 8 + v], val);   // ds_add_f32
    }
  }
  __syncthreads();

  if (tid < 16) {
    int p = p0 + tid;
    if (p < NPAIR)
      tril[(size_t)b * NPAIR + p] = tanhf(rowsum[tid] + b2[0]);
  }
}

// ---------------------------------------------------------------- batched Jacobi eigh
__global__ __launch_bounds__(256)
void eig_kernel(const float* __restrict__ tril, const float* __restrict__ vols,
                float* __restrict__ out) {
  extern __shared__ float smem[];
  float* A     = smem;                       // 100*101
  float* V     = A + NUM_ASSETS * LDA;       // 100*101
  float* cc    = V + NUM_ASSETS * LDA;       // 50
  float* ss    = cc + 50;                    // 50
  float* sq    = ss + 50;                    // 100
  float* inv_s = sq + NUM_ASSETS;            // 100
  float* vl    = inv_s + NUM_ASSETS;         // 100
  int*   pp    = (int*)(vl + NUM_ASSETS);    // 50
  int*   qq    = pp + 50;                    // 50

  int b   = blockIdx.x;
  int tid = threadIdx.x;
  const float* tb = tril + (size_t)b * NPAIR;

  for (int idx = tid; idx < NUM_ASSETS * NUM_ASSETS; idx += 256) {
    int i = idx / NUM_ASSETS, j = idx - i * NUM_ASSETS;
    float v;
    if (i == j) v = 1.f;
    else {
      int ii = (i > j) ? i : j, jj = (i > j) ? j : i;
      v = tb[ii * (ii - 1) / 2 + jj];
    }
    A[i * LDA + j] = v;
    V[i * LDA + j] = (i == j) ? 1.f : 0.f;
  }
  __syncthreads();

  for (int sweep = 0; sweep < NSWEEP; ++sweep) {
    for (int r = 0; r < 99; ++r) {
      if (tid < 50) {
        int p, q;
        if (tid == 0) { p = 0; q = 1 + (98 + r) % 99; }
        else { p = 1 + (tid - 1 + r) % 99; q = 1 + (98 - tid + r) % 99; }
        if (p > q) { int t2 = p; p = q; q = t2; }
        float app = A[p * LDA + p], aqq = A[q * LDA + q], apq = A[p * LDA + q];
        float c = 1.f, s = 0.f;
        if (fabsf(apq) > 1e-12f) {
          float tau = (aqq - app) / (2.f * apq);
          float t = (tau >= 0.f ? 1.f : -1.f) / (fabsf(tau) + sqrtf(1.f + tau * tau));
          c = rsqrtf(1.f + t * t);
          s = t * c;
        }
        pp[tid] = p; qq[tid] = q; cc[tid] = c; ss[tid] = s;
      }
      __syncthreads();
      for (int idx = tid; idx < 50 * NUM_ASSETS; idx += 256) {
        int k = idx / NUM_ASSETS, j = idx - k * NUM_ASSETS;
        int p = pp[k], q = qq[k];
        float c = cc[k], s = ss[k];
        float ap = A[p * LDA + j], aq = A[q * LDA + j];
        A[p * LDA + j] = c * ap + s * aq;
        A[q * LDA + j] = c * aq - s * ap;
      }
      __syncthreads();
      for (int idx = tid; idx < 50 * NUM_ASSETS; idx += 256) {
        int k = idx / NUM_ASSETS, i = idx - k * NUM_ASSETS;
        int p = pp[k], q = qq[k];
        float c = cc[k], s = ss[k];
        float ap = A[i * LDA + p], aq = A[i * LDA + q];
        A[i * LDA + p] = c * ap + s * aq;
        A[i * LDA + q] = c * aq - s * ap;
        float vp = V[i * LDA + p], vq = V[i * LDA + q];
        V[i * LDA + p] = c * vp + s * vq;
        V[i * LDA + q] = c * vq - s * vp;
      }
      __syncthreads();
    }
  }

  if (tid < NUM_ASSETS) {
    sq[tid] = sqrtf(fmaxf(A[tid * LDA + tid], 1e-4f));
    vl[tid] = vols[b * NUM_ASSETS + tid];
  }
  __syncthreads();
  for (int idx = tid; idx < NUM_ASSETS * NUM_ASSETS; idx += 256) {
    int i = idx / NUM_ASSETS, k = idx - i * NUM_ASSETS;
    A[i * LDA + k] = V[i * LDA + k] * sq[k];
  }
  __syncthreads();
  if (tid < NUM_ASSETS) {
    float d = 0.f;
    for (int k = 0; k < NUM_ASSETS; ++k) {
      float u = A[tid * LDA + k];
      d = fmaf(u, u, d);
    }
    inv_s[tid] = rsqrtf(fmaxf(d, 1e-6f));
  }
  __syncthreads();

  float* ob = out + (size_t)b * NUM_ASSETS * NUM_ASSETS;
  for (int idx = tid; idx < NUM_ASSETS * NUM_ASSETS; idx += 256) {
    int i = idx / NUM_ASSETS, j = idx - i * NUM_ASSETS;
    float r = 0.f;
    for (int k = 0; k < NUM_ASSETS; ++k)
      r = fmaf(A[i * LDA + k], A[j * LDA + k], r);
    ob[idx] = vl[i] * inv_s[i] * r * inv_s[j] * vl[j];
  }
}

// ---------------------------------------------------------------- launcher
extern "C" void kernel_launch(void* const* d_in, const int* in_sizes, int n_in,
                              void* d_out, int out_size, void* d_ws, size_t ws_size,
                              hipStream_t stream) {
  const float* x       = (const float*)d_in[0];
  const float* vol_w1  = (const float*)d_in[1];
  const float* vol_b1  = (const float*)d_in[2];
  const float* vol_w2  = (const float*)d_in[3];
  const float* vol_b2  = (const float*)d_in[4];
  const float* corr_w1 = (const float*)d_in[5];
  const float* corr_b1 = (const float*)d_in[6];
  const float* corr_w2 = (const float*)d_in[7];
  const float* corr_b2 = (const float*)d_in[8];
  float* out = (float*)d_out;

  char*   ws   = (char*)d_ws;
  __bf16* w1f  = (__bf16*)ws;                                        // 128 KB
  float*  tril = (float*)(ws + (size_t)D_MODEL * HIDDEN * 2);        // 64*4950*4
  float*  vols = (float*)(ws + (size_t)D_MODEL * HIDDEN * 2
                             + (size_t)BATCH * NPAIR * 4);           // 64*100*4

  cvt_w1_kernel<<<(D_MODEL * HIDDEN + 255) / 256, 256, 0, stream>>>(corr_w1, w1f);
  vol_kernel<<<BATCH * NUM_ASSETS, 256, 0, stream>>>(
      x, vol_w1, vol_b1, vol_w2, vol_b2, vols);
  corr_kernel<<<BATCH * TILES_PB, 256, 0, stream>>>(
      x, w1f, corr_b1, corr_w2, corr_b2, tril);

  size_t smem = (size_t)(2 * NUM_ASSETS * LDA + 100 + 3 * NUM_ASSETS) * sizeof(float)
              + 100 * sizeof(int);
  hipFuncSetAttribute((const void*)eig_kernel,
                      hipFuncAttributeMaxDynamicSharedMemorySize, (int)smem);
  eig_kernel<<<BATCH, 256, smem, stream>>>(tril, vols, out);
}
